// ComplexS4Model_48137993453867
// MI455X (gfx1250) — compile-verified, matrix operand order checked by reading, not run on hardware
//
#include <hip/hip_runtime.h>
#include <math.h>

typedef unsigned short us;
typedef __attribute__((ext_vector_type(16))) __bf16        v16bf;
typedef __attribute__((ext_vector_type(8)))  float         v8f;
typedef __attribute__((ext_vector_type(4)))  unsigned int  u32x4;

#define B_   16
#define L_   4096
#define DIN  257
#define DOUT 257
#define H_   256
#define NL   4
#define NS   32
#define KENC 288              // 257 padded to multiple of 32
#define NDEC 272              // 257 padded to multiple of 16
#define BL   (B_*L_)          // 65536

// ---------------- helpers ----------------

__device__ __forceinline__ us f2bf(float x){
  union { float f; unsigned u; } v; v.f = x;
  unsigned r = v.u + 0x7FFFu + ((v.u >> 16) & 1u);   // round-to-nearest-even
  return (us)(r >> 16);
}

union FragU { u32x4 q[2]; v16bf v; };

// 16-bit fragment layout (CDNA5 ISA 7.12.2): half j of lane holds
// K = kbase + j + 8*(j>=8) + 8*(lane>=16)  -> two contiguous 8-element runs.
__device__ __forceinline__ v16bf load_frag16(const us* row, int kbase, int hi){
  FragU f;
  const u32x4* p = reinterpret_cast<const u32x4*>(row + kbase + hi*8);
  f.q[0] = p[0];   // K0 .. K0+7
  f.q[1] = p[2];   // K0+16 .. K0+23
  return f.v;
}

__device__ __forceinline__ v8f wmma_bf16(v16bf a, v16bf b, v8f c){
  return __builtin_amdgcn_wmma_f32_16x16x32_bf16(false, a, false, b, (short)0, c, false, false);
}

__device__ __forceinline__ v8f vzero8(){
  v8f z;
  #pragma unroll
  for(int i=0;i<8;i++) z[i]=0.f;
  return z;
}

// ---------------- staging kernels ----------------

__global__ __launch_bounds__(256) void k_f32_to_bf16(const float* __restrict__ s,
                                                     us* __restrict__ d, int n){
  int i = blockIdx.x*256 + threadIdx.x;
  if(i < n) d[i] = f2bf(s[i]);
}

// x:(B,L,257,2) -> Xr,Xi:(BL,KENC) bf16, zero padded
__global__ __launch_bounds__(256) void k_stage_x(const float* __restrict__ x,
                                                 us* __restrict__ Xr, us* __restrict__ Xi){
  int i = blockIdx.x*256 + threadIdx.x;
  if(i >= BL*KENC) return;
  int pos = i / KENC, k = i % KENC;
  float vr = 0.f, vi = 0.f;
  if(k < DIN){
    const float* p = x + ((size_t)pos*DIN + k)*2;
    vr = p[0]; vi = p[1];
  }
  Xr[i] = f2bf(vr); Xi[i] = f2bf(vi);
}

// enc W:(256,257) -> (256,KENC) bf16, zero padded
__global__ __launch_bounds__(256) void k_stage_encw(const float* __restrict__ Wr,
                                                    const float* __restrict__ Wi,
                                                    us* __restrict__ dr, us* __restrict__ di){
  int i = blockIdx.x*256 + threadIdx.x;
  if(i >= H_*KENC) return;
  int h = i / KENC, k = i % KENC;
  float wr = 0.f, wi = 0.f;
  if(k < DIN){ wr = Wr[h*DIN + k]; wi = Wi[h*DIN + k]; }
  dr[i] = f2bf(wr); di[i] = f2bf(wi);
}

// dec W:(257,256) -> (NDEC,256) bf16, zero padded rows
__global__ __launch_bounds__(256) void k_stage_decw(const float* __restrict__ Wr,
                                                    const float* __restrict__ Wi,
                                                    us* __restrict__ dr, us* __restrict__ di){
  int i = blockIdx.x*256 + threadIdx.x;
  if(i >= NDEC*H_) return;
  int o = i / H_, k = i % H_;
  float wr = 0.f, wi = 0.f;
  if(o < DOUT){ wr = Wr[o*H_ + k]; wi = Wi[o*H_ + k]; }
  dr[i] = f2bf(wr); di[i] = f2bf(wi);
}

// (B,H,L) f32 -> (B,L,H) bf16
__global__ __launch_bounds__(256) void k_transpose_bf16(const float* __restrict__ x,
                                                        us* __restrict__ xt){
  int i = blockIdx.x*256 + threadIdx.x;
  if(i >= B_*H_*L_) return;
  int l = i % L_;
  int t = i / L_;
  int h = t % H_, b = t / H_;
  xt[((size_t)b*L_ + l)*H_ + h] = f2bf(x[i]);
}

// ---------------- encoder: out[h][pos] = Wr.Xr - Wi.Xi (+br) ; Wi.Xr + Wr.Xi (+bi) ----------------
// M = H (rows from weights), N = BL (positions), K = KENC. One wave per 16x16 tile.

__global__ __launch_bounds__(256) void k_encoder(const us* __restrict__ Wr, const us* __restrict__ Wi,
                                                 const us* __restrict__ Xr, const us* __restrict__ Xi,
                                                 const float* __restrict__ br, const float* __restrict__ bi,
                                                 float* __restrict__ xr_out, float* __restrict__ xi_out){
  int wv   = (blockIdx.x*blockDim.x + threadIdx.x) >> 5;
  int lane = threadIdx.x & 31;
  const int TN = BL/16;                       // 4096 position tiles
  int tm = wv / TN;                           // 0..15 (h tiles)
  int tn = wv % TN;
  if(tm >= H_/16) return;
  int idx = lane & 15, hi = lane >> 4;

  const us* awr = Wr + (size_t)(tm*16 + idx)*KENC;
  const us* awi = Wi + (size_t)(tm*16 + idx)*KENC;
  const us* bxr = Xr + (size_t)(tn*16 + idx)*KENC;
  const us* bxi = Xi + (size_t)(tn*16 + idx)*KENC;

  v8f a_rr = vzero8(), a_ii = vzero8(), a_ir = vzero8(), a_ri = vzero8();
  #pragma unroll 3
  for(int kk = 0; kk < KENC; kk += 32){
    v16bf fwr = load_frag16(awr, kk, hi);
    v16bf fwi = load_frag16(awi, kk, hi);
    v16bf fxr = load_frag16(bxr, kk, hi);
    v16bf fxi = load_frag16(bxi, kk, hi);
    a_rr = wmma_bf16(fwr, fxr, a_rr);   // Wr.Xr
    a_ii = wmma_bf16(fwi, fxi, a_ii);   // Wi.Xi
    a_ir = wmma_bf16(fwi, fxr, a_ir);   // Wi.Xr
    a_ri = wmma_bf16(fwr, fxi, a_ri);   // Wr.Xi
  }
  int pos = tn*16 + idx;
  int b = pos / L_, l = pos % L_;
  #pragma unroll
  for(int r = 0; r < 8; r++){
    int h = tm*16 + r + hi*8;
    size_t o = ((size_t)b*H_ + h)*L_ + l;
    xr_out[o] = a_rr[r] - a_ii[r] + br[h];
    xi_out[o] = a_ir[r] + a_ri[r] + bi[h];
  }
}

// ---------------- S4D recurrence scan + D*u + GELU, bf16 transposed output ----------------
// wave per (b,h); lane = mode n. y[l] = Re(2*C*(e^{dtA}-1)/A  *  s[l]),  s = w*s + u.

__global__ __launch_bounds__(256) void k_scan(const float* __restrict__ log_dt,
                                              const float* __restrict__ lAr,
                                              const float* __restrict__ Aim,
                                              const float* __restrict__ C2,
                                              const float* __restrict__ Dp,
                                              const float* __restrict__ u,
                                              us* __restrict__ Yt, int combo){
  __shared__ float xpose[8][32][33];
  int w = threadIdx.x >> 5, lane = threadIdx.x & 31;
  int g = blockIdx.x*8 + w;                 // 0..B*H-1
  int b = g >> 8, h = g & 255;

  float dt  = __expf(log_dt[combo*H_ + h]);
  int pidx  = (combo*H_ + h)*NS + lane;
  float Ar  = -__expf(lAr[pidx]);
  float Ai  = Aim[pidx];
  float dAr = dt*Ar, dAi = dt*Ai;
  float er  = __expf(dAr);
  float wr  = er*__cosf(dAi);
  float wi  = er*__sinf(dAi);
  float Cr  = C2[pidx*2 + 0], Ci = C2[pidx*2 + 1];
  float Emr = wr - 1.f, Emi = wi;                 // e^{dtA} - 1
  float tr_ = Cr*Emr - Ci*Emi;                    // C*(E-1)
  float ti_ = Cr*Emi + Ci*Emr;
  float den = 1.f/(Ar*Ar + Ai*Ai);
  float cmr = 2.f*(tr_*Ar + ti_*Ai)*den;          // 2*C*(E-1)/A
  float cmi = 2.f*(ti_*Ar - tr_*Ai)*den;
  float Dh  = Dp[combo*H_ + h];

  const float* up = u + (size_t)g * L_;
  us* yp = Yt + (size_t)b*L_*H_ + h;

  float sr = 0.f, si = 0.f;
  for(int tile = 0; tile < L_/32; ++tile){
    float ureg = up[tile*32 + lane];
    float c[32];
    #pragma unroll
    for(int t = 0; t < 32; ++t){
      float ut  = __shfl(ureg, t, 32);
      float nsr = fmaf(wr, sr, fmaf(-wi, si, ut));
      float nsi = fmaf(wr, si, wi*sr);
      sr = nsr; si = nsi;
      c[t] = fmaf(cmr, sr, -cmi*si);              // Re(Cm * s)
    }
    __syncthreads();
    #pragma unroll
    for(int t = 0; t < 32; ++t) xpose[w][t][lane] = c[t];
    __syncthreads();
    float acc = 0.f;
    #pragma unroll
    for(int n = 0; n < 32; ++n) acc += xpose[w][lane][n];
    float y = fmaf(Dh, ureg, acc);
    y = 0.5f * y * (1.f + erff(y * 0.70710678118654752f));   // exact GELU
    yp[(size_t)(tile*32 + lane)*H_] = f2bf(y);
  }
}

// ---------------- Wout (2H x H) GEMM over (B,L) positions + bias + GLU ----------------
// A = Wout rows (h and h+256), B[k][l] = Yt[b][l][k]. mode: 0=store 1=add 2=sub.

__global__ __launch_bounds__(256) void k_wout(const us* __restrict__ Wst,
                                              const float* __restrict__ bout_all,
                                              const us* __restrict__ Yt,
                                              float* __restrict__ out, int combo, int mode){
  int wv   = (blockIdx.x*blockDim.x + threadIdx.x) >> 5;
  int lane = threadIdx.x & 31;
  const int TPB = (H_/16)*(L_/16);              // 4096 tiles per batch
  int b  = wv / TPB;
  int t  = wv % TPB;
  int tm = t / (L_/16);                          // h tile 0..15
  int tn = t % (L_/16);                          // l tile
  if(b >= B_) return;
  int idx = lane & 15, hi = lane >> 4;

  const us* W  = Wst + (size_t)combo*512*256;
  const us* aa = W + (size_t)(tm*16 + idx)*256;         // rows h      (a half)
  const us* ag = W + (size_t)(tm*16 + idx + 256)*256;   // rows h+256  (gate half)
  const us* bb = Yt + ((size_t)b*L_ + tn*16 + idx)*H_;

  v8f acca = vzero8(), accg = vzero8();
  #pragma unroll
  for(int kk = 0; kk < 256; kk += 32){
    v16bf fb = load_frag16(bb, kk, hi);
    v16bf fa = load_frag16(aa, kk, hi);
    v16bf fg = load_frag16(ag, kk, hi);
    acca = wmma_bf16(fa, fb, acca);
    accg = wmma_bf16(fg, fb, accg);
  }
  const float* bo = bout_all + combo*512;
  int l = tn*16 + idx;
  #pragma unroll
  for(int r = 0; r < 8; r++){
    int h = tm*16 + r + hi*8;
    float a = acca[r] + bo[h];
    float g = accg[r] + bo[h + 256];
    float v = a * (1.f/(1.f + __expf(-g)));             // GLU
    size_t o = ((size_t)b*H_ + h)*L_ + l;
    if(mode == 0)      out[o]  = v;
    else if(mode == 1) out[o] += v;
    else               out[o] -= v;
  }
}

// ---------------- residual add + LayerNorm over H (per (b,l)) ----------------

__global__ __launch_bounds__(256) void k_update_ln(float* __restrict__ xr, float* __restrict__ xi,
                                                   const float* __restrict__ zr, const float* __restrict__ zi,
                                                   const float* __restrict__ gamma, const float* __restrict__ beta,
                                                   int layer){
  __shared__ float4 sh[256];
  int pos = blockIdx.x;
  int b = pos / L_, l = pos % L_;
  int h = threadIdx.x;
  size_t o = ((size_t)b*H_ + h)*L_ + l;
  float vr = xr[o] + zr[o];
  float vi = xi[o] + zi[o];
  sh[h] = make_float4(vr, vr*vr, vi, vi*vi);
  __syncthreads();
  for(int st = 128; st > 0; st >>= 1){
    if(h < st){
      float4 a = sh[h], c = sh[h + st];
      sh[h] = make_float4(a.x + c.x, a.y + c.y, a.z + c.z, a.w + c.w);
    }
    __syncthreads();
  }
  float4 s = sh[0];
  const float inv = 1.f/(float)H_;
  float mr = s.x*inv, vvr = s.y*inv - mr*mr;
  float mi = s.z*inv, vvi = s.w*inv - mi*mi;
  float gr = gamma[(layer*2 + 0)*H_ + h], br = beta[(layer*2 + 0)*H_ + h];
  float gi = gamma[(layer*2 + 1)*H_ + h], bi = beta[(layer*2 + 1)*H_ + h];
  xr[o] = (vr - mr)*rsqrtf(vvr + 1e-5f)*gr + br;
  xi[o] = (vi - mi)*rsqrtf(vvi + 1e-5f)*gi + bi;
}

// ---------------- decoder: out[pos][o] = Xr.Wr - Xi.Wi ; Xr.Wi + Xi.Wr ----------------
// M = BL (positions), N = NDEC (outputs), K = 256.

__global__ __launch_bounds__(256) void k_decoder(const us* __restrict__ Xrt, const us* __restrict__ Xit,
                                                 const us* __restrict__ Wr, const us* __restrict__ Wi,
                                                 const float* __restrict__ br, const float* __restrict__ bi,
                                                 float* __restrict__ out){
  int wv   = (blockIdx.x*blockDim.x + threadIdx.x) >> 5;
  int lane = threadIdx.x & 31;
  const int TN = NDEC/16;                        // 17
  int tm = wv / TN;
  int tn = wv % TN;
  if(tm >= BL/16) return;
  int idx = lane & 15, hi = lane >> 4;

  const us* ar = Xrt + (size_t)(tm*16 + idx)*H_;
  const us* ai = Xit + (size_t)(tm*16 + idx)*H_;
  const us* wrp = Wr + (size_t)(tn*16 + idx)*H_;
  const us* wip = Wi + (size_t)(tn*16 + idx)*H_;

  v8f a_rr = vzero8(), a_ii = vzero8(), a_ri = vzero8(), a_ir = vzero8();
  #pragma unroll
  for(int kk = 0; kk < 256; kk += 32){
    v16bf fxr = load_frag16(ar, kk, hi);
    v16bf fxi = load_frag16(ai, kk, hi);
    v16bf fwr = load_frag16(wrp, kk, hi);
    v16bf fwi = load_frag16(wip, kk, hi);
    a_rr = wmma_bf16(fxr, fwr, a_rr);
    a_ii = wmma_bf16(fxi, fwi, a_ii);
    a_ri = wmma_bf16(fxr, fwi, a_ri);
    a_ir = wmma_bf16(fxi, fwr, a_ir);
  }
  int oc = tn*16 + idx;
  if(oc < DOUT){
    #pragma unroll
    for(int r = 0; r < 8; r++){
      int pos = tm*16 + r + hi*8;
      size_t base = ((size_t)pos*DOUT + oc)*2;
      out[base + 0] = a_rr[r] - a_ii[r] + br[oc];
      out[base + 1] = a_ri[r] + a_ir[r] + bi[oc];
    }
  }
}

// ---------------- host orchestration ----------------

extern "C" void kernel_launch(void* const* d_in, const int* in_sizes, int n_in,
                              void* d_out, int out_size, void* d_ws, size_t ws_size,
                              hipStream_t stream) {
  const float* x        = (const float*)d_in[0];
  const float* enc_Wr   = (const float*)d_in[1];
  const float* enc_Wi   = (const float*)d_in[2];
  const float* enc_br   = (const float*)d_in[3];
  const float* enc_bi   = (const float*)d_in[4];
  const float* s4_logdt = (const float*)d_in[5];
  const float* s4_lAr   = (const float*)d_in[6];
  const float* s4_Aim   = (const float*)d_in[7];
  const float* s4_C     = (const float*)d_in[8];
  const float* s4_D     = (const float*)d_in[9];
  const float* s4_Wout  = (const float*)d_in[10];
  const float* s4_bout  = (const float*)d_in[11];
  const float* ln_g     = (const float*)d_in[12];
  const float* ln_b     = (const float*)d_in[13];
  const float* dec_Wr   = (const float*)d_in[14];
  const float* dec_Wi   = (const float*)d_in[15];
  const float* dec_br   = (const float*)d_in[16];
  const float* dec_bi   = (const float*)d_in[17];
  float* out = (float*)d_out;

  char* p = (char*)d_ws;
  auto carve = [&](size_t bytes) -> void* {
    void* r = (void*)p;
    p += (bytes + 255) & ~(size_t)255;
    return r;
  };
  const size_t act = (size_t)B_*H_*L_;
  float* xr = (float*)carve(act*4);
  float* xi = (float*)carve(act*4);
  float* zr = (float*)carve(act*4);
  float* zi = (float*)carve(act*4);
  us* Yt    = (us*)carve(act*2);        // (B,L,H) bf16
  us* Yt2   = (us*)carve(act*2);
  us* Xr_s  = (us*)carve((size_t)BL*KENC*2);
  us* Xi_s  = (us*)carve((size_t)BL*KENC*2);
  us* eWr   = (us*)carve((size_t)H_*KENC*2);
  us* eWi   = (us*)carve((size_t)H_*KENC*2);
  us* dWr   = (us*)carve((size_t)NDEC*H_*2);
  us* dWi   = (us*)carve((size_t)NDEC*H_*2);
  us* WoutS = (us*)carve((size_t)8*512*256*2);

  // --- staging ---
  k_stage_x   <<<(BL*KENC + 255)/256, 256, 0, stream>>>(x, Xr_s, Xi_s);
  k_stage_encw<<<(H_*KENC + 255)/256, 256, 0, stream>>>(enc_Wr, enc_Wi, eWr, eWi);
  k_stage_decw<<<(NDEC*H_ + 255)/256, 256, 0, stream>>>(dec_Wr, dec_Wi, dWr, dWi);
  k_f32_to_bf16<<<(8*512*256 + 255)/256, 256, 0, stream>>>(s4_Wout, WoutS, 8*512*256);

  // --- encoder ---
  {
    int waves = (H_/16)*(BL/16);             // 65536
    k_encoder<<<waves*32/256, 256, 0, stream>>>(eWr, eWi, Xr_s, Xi_s, enc_br, enc_bi, xr, xi);
  }

  // --- layers ---
  int gw = ((H_/16)*(L_/16)*B_)*32/256;       // 8192 blocks for k_wout
  for(int l = 0; l < NL; ++l){
    int c0 = l*2, c1 = l*2 + 1;
    // zr = b0(xr)
    k_scan<<<512, 256, 0, stream>>>(s4_logdt, s4_lAr, s4_Aim, s4_C, s4_D, xr, Yt, c0);
    k_wout<<<gw, 256, 0, stream>>>(WoutS, s4_bout, Yt, zr, c0, 0);
    // zi = b0(xi)
    k_scan<<<512, 256, 0, stream>>>(s4_logdt, s4_lAr, s4_Aim, s4_C, s4_D, xi, Yt, c0);
    k_wout<<<gw, 256, 0, stream>>>(WoutS, s4_bout, Yt, zi, c0, 0);
    // zi += b1(xr)
    k_scan<<<512, 256, 0, stream>>>(s4_logdt, s4_lAr, s4_Aim, s4_C, s4_D, xr, Yt, c1);
    k_wout<<<gw, 256, 0, stream>>>(WoutS, s4_bout, Yt, zi, c1, 1);
    // zr -= b1(xi)
    k_scan<<<512, 256, 0, stream>>>(s4_logdt, s4_lAr, s4_Aim, s4_C, s4_D, xi, Yt, c1);
    k_wout<<<gw, 256, 0, stream>>>(WoutS, s4_bout, Yt, zr, c1, 2);
    // residual + post-LN
    k_update_ln<<<BL, 256, 0, stream>>>(xr, xi, zr, zi, ln_g, ln_b, l);
  }

  // --- decoder ---
  k_transpose_bf16<<<(int)((act + 255)/256), 256, 0, stream>>>(xr, Yt);
  k_transpose_bf16<<<(int)((act + 255)/256), 256, 0, stream>>>(xi, Yt2);
  {
    int waves = (BL/16)*(NDEC/16);           // 69632
    k_decoder<<<waves*32/256, 256, 0, stream>>>(Yt, Yt2, dWr, dWi, dec_br, dec_bi, out);
  }
}
